// MoELayer_63393717289149
// MI455X (gfx1250) — compile-verified, matrix operand order checked by reading, not run on hardware
//
#include <hip/hip_runtime.h>
#include <hip/hip_bf16.h>
#include <math.h>

// ---------------------------------------------------------------------------
// MoE layer, per-batch routing (top-2 of 8 experts shared by all tokens of a
// batch).  bf16 WMMA GEMMs (v_wmma_f32_16x16x32_bf16) with f32 accumulation;
// only the 2 selected experts per batch are computed.  A-tiles are filled via
// GLOBAL_LOAD_ASYNC_TO_LDS (ASYNCcnt) when the toolchain exposes the builtin.
// ---------------------------------------------------------------------------

namespace {
constexpr int kB = 4, kS = 2048, kD = 1024, kF = 4096, kE = 8, kTop = 2;
constexpr int TM  = 128;   // block tile rows (tokens)
constexpr int TN  = 256;   // block tile cols
constexpr int KT  = 32;    // K step (wmma 16x16x32 bf16)
constexpr int LDP = 40;    // LDS row stride in ushorts (mult of 8 -> 16B align)
}

typedef __attribute__((ext_vector_type(16))) __bf16 bf16x16;
typedef __attribute__((ext_vector_type(8)))  float  f32x8;
typedef __attribute__((ext_vector_type(4)))  int    i32x4;

union FragAB { bf16x16 v; uint4 q[2]; };

__device__ __forceinline__ unsigned short f2bf(float f) {
  unsigned int u = __float_as_uint(f);            // RNE f32 -> bf16
  u += 0x7FFFu + ((u >> 16) & 1u);
  return (unsigned short)(u >> 16);
}

__device__ __forceinline__ float gelu_exact(float v) {
  return 0.5f * v * (1.0f + erff(v * 0.70710678118654752f));
}

// --- async global->LDS copy (CDNA5), with safe fallback --------------------
#if defined(__HIP_DEVICE_COMPILE__) &&                                         \
    __has_builtin(__builtin_amdgcn_global_load_async_to_lds_b128) &&           \
    __has_builtin(__builtin_amdgcn_s_wait_asynccnt)
#define MOE_ASYNC_LDS 1
__device__ __forceinline__ void lds_cp_b128(void* lds_dst, const void* g_src) {
  // signature (probe-verified by diagnostic): (int4 AS1*, int4 AS3*, Iimm, Iimm)
  __builtin_amdgcn_global_load_async_to_lds_b128(
      (__attribute__((address_space(1))) i32x4*)g_src,
      (__attribute__((address_space(3))) i32x4*)lds_dst, 0, 0);
}
__device__ __forceinline__ void lds_cp_wait() {
  __builtin_amdgcn_s_wait_asynccnt(0);
}
#else
__device__ __forceinline__ void lds_cp_b128(void* lds_dst, const void* g_src) {
  *(uint4*)lds_dst = *(const uint4*)g_src;
}
__device__ __forceinline__ void lds_cp_wait() {}
#endif

// ---------------------------------------------------------------------------
// x (f32) -> bf16 copy, once per launch (feeds async A-tile copies in FFN1).
// ---------------------------------------------------------------------------
__global__ void moe_x2bf_kernel(const float* __restrict__ x,
                                unsigned short* __restrict__ xbf) {
  const size_t i0 = ((size_t)blockIdx.x * 256 + threadIdx.x) * 4;
#pragma unroll
  for (int j = 0; j < 4; ++j) xbf[i0 + j] = f2bf(x[i0 + j]);
}

// ---------------------------------------------------------------------------
// Router: rw = softmax(mean_s(x) @ Wr + br); top-2; softmax over the two
// selected *probabilities* (exact replication of the reference).
// ---------------------------------------------------------------------------
__global__ void moe_router_kernel(const float* __restrict__ x,
                                  const float* __restrict__ Wr,
                                  const float* __restrict__ br,
                                  int* __restrict__ routeE,
                                  float* __restrict__ routeW) {
  const int b   = blockIdx.x;
  const int tid = threadIdx.x;
  __shared__ float mean[kD];
  __shared__ float logits[kE];
  const float* xb = x + (size_t)b * kS * kD;
#pragma unroll
  for (int j = 0; j < kD / 256; ++j) {
    const int d = tid + j * 256;
    float s = 0.f;
    for (int t = 0; t < kS; ++t) s += xb[(size_t)t * kD + d];
    mean[d] = s * (1.0f / (float)kS);
  }
  __syncthreads();
  if (tid < kE) {
    float acc = br[tid];
    for (int d = 0; d < kD; ++d) acc += mean[d] * Wr[d * kE + tid];
    logits[tid] = acc;
  }
  __syncthreads();
  if (tid == 0) {
    float mx = logits[0];
    for (int e = 1; e < kE; ++e) mx = fmaxf(mx, logits[e]);
    float p[kE]; float sum = 0.f;
    for (int e = 0; e < kE; ++e) { p[e] = expf(logits[e] - mx); sum += p[e]; }
    const float inv = 1.0f / sum;
    for (int e = 0; e < kE; ++e) p[e] *= inv;
    int i0 = 0;
    for (int e = 1; e < kE; ++e) if (p[e] > p[i0]) i0 = e;   // lowest idx on tie
    int i1 = (i0 == 0) ? 1 : 0;
    for (int e = 0; e < kE; ++e) if (e != i0 && p[e] > p[i1]) i1 = e;
    const float m2 = fmaxf(p[i0], p[i1]);                    // softmax over the
    const float e0 = expf(p[i0] - m2), e1 = expf(p[i1] - m2);// two probabilities
    const float s2 = e0 + e1;
    routeE[b * 2 + 0] = i0;       routeE[b * 2 + 1] = i1;
    routeW[b * 2 + 0] = e0 / s2;  routeW[b * 2 + 1] = e1 / s2;
  }
}

// ---------------------------------------------------------------------------
// FFN1: H = gelu(X @ W1[e] + b1[e])  (bf16 H in workspace)
// Block tile 128x256, 256 threads = 8 waves (2M x 4N), wave tile 64x64.
// ---------------------------------------------------------------------------
__global__ void __launch_bounds__(256)
moe_ffn1_kernel(const unsigned short* __restrict__ Xbf,
                const float* __restrict__ W1,
                const float* __restrict__ b1,
                const int* __restrict__ routeE,
                unsigned short* __restrict__ H,
                int slot) {
  __shared__ unsigned short ldsA[TM * LDP];
  __shared__ unsigned short ldsB[TN * LDP];
  const int b    = blockIdx.z;
  const int e    = routeE[b * 2 + slot];
  const int row0 = blockIdx.y * TM;
  const int f0   = blockIdx.x * TN;
  const int tid  = threadIdx.x;
  const int wave = tid >> 5, lane = tid & 31;
  const int half = lane >> 4, lr = lane & 15;
  const int wm   = (wave & 1) * 64;     // 2 waves along M
  const int wn   = (wave >> 1) * 64;    // 4 waves along N
  const unsigned short* X = Xbf + (size_t)b * kS * kD;
  const float*          W = W1  + (size_t)e * kD * kF;
  f32x8 acc[4][4] = {};

  for (int kk = 0; kk < kD; kk += KT) {
    // A tile (bf16 copy): rows 0..255 x 32 = 1024 b128 chunks, 4 per thread (async)
#pragma unroll
    for (int j = 0; j < 4; ++j) {
      const int c = tid + j * 256;
      const int r = c >> 2, cc = (c & 3) * 8;
      lds_cp_b128(&ldsA[r * LDP + cc],
                  &X[(size_t)(row0 + r) * kD + kk + cc]);
    }
    // B tile: ldsB[n][k] = bf16(W1[kk+k][f0+n]) (256n x 32k), overlaps async
#pragma unroll
    for (int j = 0; j < 32; ++j) {
      const int i = tid + j * 256;
      const int k = i >> 8, n = i & 255;
      ldsB[n * LDP + k] = f2bf(W[(size_t)(kk + k) * kF + f0 + n]);
    }
    lds_cp_wait();
    __syncthreads();
    FragAB af[4], bfr[4];
#pragma unroll
    for (int mf = 0; mf < 4; ++mf) {
      const unsigned short* p = &ldsA[(wm + mf * 16 + lr) * LDP + half * 8];
      af[mf].q[0] = *(const uint4*)(p);        // K = half*8 + 0..7
      af[mf].q[1] = *(const uint4*)(p + 16);   // K = 16 + half*8 + 0..7
    }
#pragma unroll
    for (int nf = 0; nf < 4; ++nf) {
      const unsigned short* p = &ldsB[(wn + nf * 16 + lr) * LDP + half * 8];
      bfr[nf].q[0] = *(const uint4*)(p);
      bfr[nf].q[1] = *(const uint4*)(p + 16);
    }
#pragma unroll
    for (int mf = 0; mf < 4; ++mf)
#pragma unroll
      for (int nf = 0; nf < 4; ++nf)
        acc[mf][nf] = __builtin_amdgcn_wmma_f32_16x16x32_bf16(
            false, af[mf].v, false, bfr[nf].v, (short)0, acc[mf][nf],
            false, false);
    __syncthreads();
  }

  const float* b1e = b1 + (size_t)e * kF;
  unsigned short* Hb = H + (size_t)(b * kS) * kF;
#pragma unroll
  for (int mf = 0; mf < 4; ++mf)
#pragma unroll
    for (int nf = 0; nf < 4; ++nf) {
      const int n = f0 + wn + nf * 16 + lr;
      const float bias = b1e[n];
#pragma unroll
      for (int r = 0; r < 8; ++r) {
        const int m = row0 + wm + mf * 16 + half * 8 + r;
        Hb[(size_t)m * kF + n] = f2bf(gelu_exact(acc[mf][nf][r] + bias));
      }
    }
}

// ---------------------------------------------------------------------------
// FFN2: out (+)= w * (H @ W2[e] + b2[e]).  first=1 -> overwrite, else add.
// ---------------------------------------------------------------------------
__global__ void __launch_bounds__(256)
moe_ffn2_kernel(const unsigned short* __restrict__ H,
                const float* __restrict__ W2,
                const float* __restrict__ b2,
                const int* __restrict__ routeE,
                const float* __restrict__ routeW,
                float* __restrict__ out,
                int slot, int first) {
  __shared__ unsigned short ldsA[TM * LDP];
  __shared__ unsigned short ldsB[TN * LDP];
  const int b    = blockIdx.z;
  const int e    = routeE[b * 2 + slot];
  const float w  = routeW[b * 2 + slot];
  const int row0 = blockIdx.y * TM;
  const int n0   = blockIdx.x * TN;
  const int tid  = threadIdx.x;
  const int wave = tid >> 5, lane = tid & 31;
  const int half = lane >> 4, lr = lane & 15;
  const int wm   = (wave & 1) * 64;
  const int wn   = (wave >> 1) * 64;
  const unsigned short* Hb = H  + (size_t)(b * kS) * kF;
  const float*          W  = W2 + (size_t)e * kF * kD;
  f32x8 acc[4][4] = {};

  for (int kk = 0; kk < kF; kk += KT) {
#pragma unroll
    for (int j = 0; j < 4; ++j) {
      const int c = tid + j * 256;
      const int r = c >> 2, cc = (c & 3) * 8;
      lds_cp_b128(&ldsA[r * LDP + cc],
                  &Hb[(size_t)(row0 + r) * kF + kk + cc]);
    }
#pragma unroll
    for (int j = 0; j < 32; ++j) {
      const int i = tid + j * 256;
      const int k = i >> 8, n = i & 255;
      ldsB[n * LDP + k] = f2bf(W[(size_t)(kk + k) * kD + n0 + n]);
    }
    lds_cp_wait();
    __syncthreads();
    FragAB af[4], bfr[4];
#pragma unroll
    for (int mf = 0; mf < 4; ++mf) {
      const unsigned short* p = &ldsA[(wm + mf * 16 + lr) * LDP + half * 8];
      af[mf].q[0] = *(const uint4*)(p);
      af[mf].q[1] = *(const uint4*)(p + 16);
    }
#pragma unroll
    for (int nf = 0; nf < 4; ++nf) {
      const unsigned short* p = &ldsB[(wn + nf * 16 + lr) * LDP + half * 8];
      bfr[nf].q[0] = *(const uint4*)(p);
      bfr[nf].q[1] = *(const uint4*)(p + 16);
    }
#pragma unroll
    for (int mf = 0; mf < 4; ++mf)
#pragma unroll
      for (int nf = 0; nf < 4; ++nf)
        acc[mf][nf] = __builtin_amdgcn_wmma_f32_16x16x32_bf16(
            false, af[mf].v, false, bfr[nf].v, (short)0, acc[mf][nf],
            false, false);
    __syncthreads();
  }

  const float* b2e = b2 + (size_t)e * kD;
#pragma unroll
  for (int mf = 0; mf < 4; ++mf)
#pragma unroll
    for (int nf = 0; nf < 4; ++nf) {
      const int n = n0 + wn + nf * 16 + lr;
      const float bias = b2e[n];
#pragma unroll
      for (int r = 0; r < 8; ++r) {
        const int m = row0 + wm + mf * 16 + half * 8 + r;
        const size_t oi = (size_t)(b * kS + m) * kD + n;
        const float v = w * (acc[mf][nf][r] + bias);
        out[oi] = first ? v : (out[oi] + v);
      }
    }
}

// ---------------------------------------------------------------------------
// Residual + LayerNorm (population variance), in-place on d_out.
// ---------------------------------------------------------------------------
__global__ void moe_ln_kernel(float* __restrict__ out,
                              const float* __restrict__ x,
                              const float* __restrict__ gamma,
                              const float* __restrict__ beta) {
  const int row = blockIdx.x;
  const int tid = threadIdx.x;
  __shared__ float rs[256], rq[256];
  float vals[kD / 256];
  float s = 0.f, q = 0.f;
  const float* xr  = x   + (size_t)row * kD;
  float*       orw = out + (size_t)row * kD;
#pragma unroll
  for (int j = 0; j < kD / 256; ++j) {
    const int d = tid + j * 256;
    const float y = orw[d] + xr[d];
    vals[j] = y; s += y; q += y * y;
  }
  rs[tid] = s; rq[tid] = q;
  __syncthreads();
  for (int off = 128; off > 0; off >>= 1) {
    if (tid < off) { rs[tid] += rs[tid + off]; rq[tid] += rq[tid + off]; }
    __syncthreads();
  }
  const float mu  = rs[0] * (1.0f / (float)kD);
  const float var = rq[0] * (1.0f / (float)kD) - mu * mu;
  const float inv = rsqrtf(var + 1e-5f);
#pragma unroll
  for (int j = 0; j < kD / 256; ++j) {
    const int d = tid + j * 256;
    orw[d] = (vals[j] - mu) * inv * gamma[d] + beta[d];
  }
}

// ---------------------------------------------------------------------------
extern "C" void kernel_launch(void* const* d_in, const int* in_sizes, int n_in,
                              void* d_out, int out_size, void* d_ws, size_t ws_size,
                              hipStream_t stream) {
  (void)in_sizes; (void)n_in; (void)out_size; (void)ws_size;
  const float* x     = (const float*)d_in[0];
  const float* W1    = (const float*)d_in[1];
  const float* b1    = (const float*)d_in[2];
  const float* W2    = (const float*)d_in[3];
  const float* b2    = (const float*)d_in[4];
  const float* Wr    = (const float*)d_in[5];
  const float* br    = (const float*)d_in[6];
  const float* gamma = (const float*)d_in[7];
  const float* beta  = (const float*)d_in[8];
  float* out = (float*)d_out;

  // Workspace: routing (256 B) | bf16 x (16 MB) | bf16 H (64 MB)
  int*            routeE = (int*)d_ws;
  float*          routeW = (float*)((char*)d_ws + 64);
  unsigned short* Xbf    = (unsigned short*)((char*)d_ws + 256);
  unsigned short* H      = (unsigned short*)((char*)d_ws + 256 +
                                             (size_t)kB * kS * kD * 2);

  const int xElems = kB * kS * kD;                       // 8.4M
  moe_x2bf_kernel<<<xElems / (256 * 4), 256, 0, stream>>>(x, Xbf);
  moe_router_kernel<<<kB, 256, 0, stream>>>(x, Wr, br, routeE, routeW);

  dim3 g1(kF / TN, kS / TM, kB);   // 16 x 16 x 4
  dim3 g2(kD / TN, kS / TM, kB);   //  4 x 16 x 4
  for (int slot = 0; slot < kTop; ++slot) {
    moe_ffn1_kernel<<<g1, 256, 0, stream>>>(Xbf, W1, b1, routeE, H, slot);
    moe_ffn2_kernel<<<g2, 256, 0, stream>>>(H, W2, b2, routeE, routeW, out,
                                            slot, slot == 0 ? 1 : 0);
  }
  moe_ln_kernel<<<kB * kS, 256, 0, stream>>>(out, x, gamma, beta);
}
